// GATLayer_75539884802467
// MI455X (gfx1250) — matrix operand order repaired and checked by reading, not compiled, not run on hardware
//
#include <hip/hip_runtime.h>
#include <hip/hip_bf16.h>

#define NN   50000      // nodes
#define NE   800000     // edges
#define INF_ 256        // in features
#define HDN  4          // heads
#define DD   64         // per-head out features
#define OUTF 256        // HDN*DD

typedef __attribute__((ext_vector_type(16))) __bf16 v16bf;
typedef __attribute__((ext_vector_type(8)))  float  v8f;

// ---------------- helpers ----------------
__device__ __forceinline__ unsigned short f2bf_rne(float f) {
    unsigned u = __float_as_uint(f);
    unsigned rounding = 0x7fffu + ((u >> 16) & 1u);
    return (unsigned short)((u + rounding) >> 16);
}
// monotonic float -> uint key for integer atomicMax (key 0 == -inf identity)
__device__ __forceinline__ unsigned f2key(float f) {
    unsigned b = __float_as_uint(f);
    return (b & 0x80000000u) ? ~b : (b | 0x80000000u);
}
__device__ __forceinline__ float key2f(unsigned k) {
    return __uint_as_float((k & 0x80000000u) ? (k ^ 0x80000000u) : ~k);
}

// ---------------- prep: f32 -> bf16 ----------------
__global__ void cvt_h_bf16(const float* __restrict__ h, unsigned short* __restrict__ hbf) {
    int i = blockIdx.x * 256 + threadIdx.x;           // one float4 per thread
    if (i >= NN * INF_ / 4) return;
    float4 x = ((const float4*)h)[i];
    union { unsigned short s[4]; uint2 u; } o;
    o.s[0] = f2bf_rne(x.x); o.s[1] = f2bf_rne(x.y);
    o.s[2] = f2bf_rne(x.z); o.s[3] = f2bf_rne(x.w);
    ((uint2*)hbf)[i] = o.u;
}

// transpose + convert W[k][n] -> WT[n][k] (bf16); tiny (256x256)
__global__ void cvt_wt_bf16(const float* __restrict__ W, unsigned short* __restrict__ WT) {
    int i = blockIdx.x * 256 + threadIdx.x;
    if (i >= INF_ * OUTF) return;
    int k = i >> 8, n = i & 255;
    WT[n * INF_ + k] = f2bf_rne(W[k * OUTF + n]);
}

// ---------------- WMMA GEMM: hp[N,256] = hbf[N,256] x W[256,256] ----------------
// One block (4 waves) per 16-row M tile; each wave computes 16x64 (4 N-tiles)
// with 4 accumulators, reusing the A fragment across 4 WMMAs per K-step.
__global__ void __launch_bounds__(128) gemm_hp(const unsigned short* __restrict__ A,
                                               const unsigned short* __restrict__ BT,
                                               float* __restrict__ C) {
    const int lane = threadIdx.x & 31;
    const int wave = threadIdx.x >> 5;        // 0..3 -> 64-column group
    const int tm   = blockIdx.x;              // 0..3124
    const int l16  = lane & 15;
    const int kg   = lane >> 4;               // half-wave K group

    const unsigned short* arow = A + (tm * 16 + l16) * INF_ + kg * 16;
    const int n0 = wave * 64 + l16;
    const unsigned short* brow0 = BT + (n0 +  0) * INF_ + kg * 16;
    const unsigned short* brow1 = BT + (n0 + 16) * INF_ + kg * 16;
    const unsigned short* brow2 = BT + (n0 + 32) * INF_ + kg * 16;
    const unsigned short* brow3 = BT + (n0 + 48) * INF_ + kg * 16;

    v8f acc0 = {}, acc1 = {}, acc2 = {}, acc3 = {};
#pragma unroll
    for (int kk = 0; kk < INF_; kk += 32) {
        union { v16bf v; uint4 q[2]; } af, b0, b1, b2, b3;
        af.q[0] = *(const uint4*)(arow + kk);        // K = kk+kg*16   .. +7
        af.q[1] = *(const uint4*)(arow + kk + 8);    // K = kk+kg*16+8 .. +15
        b0.q[0] = *(const uint4*)(brow0 + kk);
        b0.q[1] = *(const uint4*)(brow0 + kk + 8);
        b1.q[0] = *(const uint4*)(brow1 + kk);
        b1.q[1] = *(const uint4*)(brow1 + kk + 8);
        b2.q[0] = *(const uint4*)(brow2 + kk);
        b2.q[1] = *(const uint4*)(brow2 + kk + 8);
        b3.q[0] = *(const uint4*)(brow3 + kk);
        b3.q[1] = *(const uint4*)(brow3 + kk + 8);
        acc0 = __builtin_amdgcn_wmma_f32_16x16x32_bf16(false, af.v, false, b0.v,
                                                       (short)0, acc0, false, false);
        acc1 = __builtin_amdgcn_wmma_f32_16x16x32_bf16(false, af.v, false, b1.v,
                                                       (short)0, acc1, false, false);
        acc2 = __builtin_amdgcn_wmma_f32_16x16x32_bf16(false, af.v, false, b2.v,
                                                       (short)0, acc2, false, false);
        acc3 = __builtin_amdgcn_wmma_f32_16x16x32_bf16(false, af.v, false, b3.v,
                                                       (short)0, acc3, false, false);
    }
    // D layout: lane l, VGPR r -> M = r + 8*(l>>4), N = l&15
    float* cbase = C + (tm * 16 + 8 * kg) * OUTF + wave * 64 + l16;
#pragma unroll
    for (int r = 0; r < 8; ++r) {
        cbase[r * OUTF +  0] = acc0[r];
        cbase[r * OUTF + 16] = acc1[r];
        cbase[r * OUTF + 32] = acc2[r];
        cbase[r * OUTF + 48] = acc3[r];
    }
}

// ---------------- per-node attention scores ----------------
__global__ void node_scores(const float* __restrict__ hp,
                            const float* __restrict__ a_src,
                            const float* __restrict__ a_dst,
                            float* __restrict__ ssrc, float* __restrict__ sdst) {
    int i = blockIdx.x * 256 + threadIdx.x;
    if (i >= NN * HDN) return;
    int n = i >> 2, h = i & 3;
    const float4* x  = (const float4*)(hp + n * OUTF + h * DD);
    const float4* as = (const float4*)(a_src + h * DD);
    const float4* ad = (const float4*)(a_dst + h * DD);
    float s1 = 0.f, s2 = 0.f;
#pragma unroll
    for (int j = 0; j < 16; ++j) {
        float4 v = x[j], p = as[j], q = ad[j];
        s1 += v.x * p.x + v.y * p.y + v.z * p.z + v.w * p.w;
        s2 += v.x * q.x + v.y * q.y + v.z * q.z + v.w * q.w;
    }
    ssrc[i] = s1; sdst[i] = s2;
}

// ---------------- init out / emax / esum ----------------
__global__ void init_bufs(float* __restrict__ out, unsigned* __restrict__ emax,
                          float* __restrict__ esum) {
    int i = blockIdx.x * 256 + threadIdx.x;
    if (i < NN * OUTF) out[i] = 0.f;
    if (i < NN * HDN) { emax[i] = 0u; esum[i] = 0.f; }
}

// ---------------- edge logits + segment max ----------------
__global__ void edge_logits(const int* __restrict__ src, const int* __restrict__ dst,
                            const float* __restrict__ ssrc, const float* __restrict__ sdst,
                            float* __restrict__ elog, unsigned* __restrict__ emax) {
    int i = blockIdx.x * 256 + threadIdx.x;
    if (i >= NE * HDN) return;
    int e = i >> 2, h = i & 3;
    int s = src[e], d = dst[e];
    float v = ssrc[s * HDN + h] + sdst[d * HDN + h];
    v = (v > 0.f) ? v : 0.2f * v;                  // leaky_relu(0.2)
    elog[i] = v;
    atomicMax(emax + d * HDN + h, f2key(v));
}

// ---------------- exp + segment sum ----------------
__global__ void edge_exp(const int* __restrict__ dst, const unsigned* __restrict__ emax,
                         float* __restrict__ elog, float* __restrict__ esum) {
    int i = blockIdx.x * 256 + threadIdx.x;
    if (i >= NE * HDN) return;
    int e = i >> 2, h = i & 3;
    int d = dst[e];
    float m  = key2f(emax[d * HDN + h]);
    float ex = __expf(elog[i] - m);
    elog[i] = ex;                                   // reuse buffer for exp values
    unsafeAtomicAdd(esum + d * HDN + h, ex);        // native global_atomic_add_f32 (L2)
}

// ---------------- weighted aggregation ----------------
__global__ void __launch_bounds__(256) aggregate(const int* __restrict__ src,
                                                 const int* __restrict__ dst,
                                                 const float* __restrict__ hp,
                                                 const float* __restrict__ elog,
                                                 const float* __restrict__ esum,
                                                 float* __restrict__ out) {
    int e = blockIdx.x;                             // one edge per block (256 thr = 4 heads x 64)
    int s = src[e], d0 = dst[e];
    int h = threadIdx.x >> 6, dd = threadIdx.x & 63;
    float alpha = elog[e * HDN + h] / (esum[d0 * HDN + h] + 1e-16f);
    unsafeAtomicAdd(out + d0 * OUTF + h * DD + dd, alpha * hp[s * OUTF + h * DD + dd]);
}

// ---------------- launch ----------------
extern "C" void kernel_launch(void* const* d_in, const int* in_sizes, int n_in,
                              void* d_out, int out_size, void* d_ws, size_t ws_size,
                              hipStream_t stream) {
    const float* h     = (const float*)d_in[0];
    const int*   ei    = (const int*)d_in[1];
    const float* W     = (const float*)d_in[2];
    const float* a_src = (const float*)d_in[3];
    const float* a_dst = (const float*)d_in[4];
    float* out = (float*)d_out;

    char* ws = (char*)d_ws;
    unsigned short* hbf = (unsigned short*)(ws + 0);          // 25,600,000 B
    unsigned short* WT  = (unsigned short*)(ws + 25600000);   //    131,072 B
    float*    hp   = (float*)(ws + 25731072);                 // 51,200,000 B
    float*    ssrc = (float*)(ws + 76931072);                 //    800,000 B
    float*    sdst = (float*)(ws + 77731072);                 //    800,000 B
    float*    elog = (float*)(ws + 78531072);                 // 12,800,000 B
    unsigned* emax = (unsigned*)(ws + 91331072);              //    800,000 B
    float*    esum = (float*)(ws + 92131072);                 //    800,000 B  (total ~93 MB)

    const int* src = ei;
    const int* dst = ei + NE;

    cvt_h_bf16 <<<(NN * INF_ / 4 + 255) / 256, 256, 0, stream>>>(h, hbf);
    cvt_wt_bf16<<<(INF_ * OUTF + 255) / 256,   256, 0, stream>>>(W, WT);

    gemm_hp<<<NN / 16, 128, 0, stream>>>(hbf, WT, hp);        // 3125 blocks x 4 waves

    node_scores<<<(NN * HDN + 255) / 256, 256, 0, stream>>>(hp, a_src, a_dst, ssrc, sdst);
    init_bufs  <<<(NN * OUTF + 255) / 256, 256, 0, stream>>>(out, emax, esum);

    edge_logits<<<(NE * HDN + 255) / 256, 256, 0, stream>>>(src, dst, ssrc, sdst, elog, emax);
    edge_exp   <<<(NE * HDN + 255) / 256, 256, 0, stream>>>(dst, emax, elog, esum);

    aggregate<<<NE, 256, 0, stream>>>(src, dst, hp, elog, esum, out);
}
